// ContextPeftAdaptorLora_47545287966875
// MI455X (gfx1250) — matrix operand order repaired and checked by reading, not compiled, not run on hardware
//
#include <hip/hip_runtime.h>
#include <hip/hip_bf16.h>

typedef __attribute__((ext_vector_type(16))) __bf16 v16bf;
typedef __attribute__((ext_vector_type(8)))  float  v8f;
typedef __attribute__((ext_vector_type(4)))  unsigned int v4u;
typedef __attribute__((ext_vector_type(8)))  int v8i;
typedef __attribute__((ext_vector_type(4)))  int v4i;

#define D_IN_   1024
#define D_OUT_  1024
#define N_ADAPT_ 4
#define R_LORA_ 16
#define NR_     64          // N_ADAPT * R
#define M_TOT_  65536       // B * T
#define SCALING_ 2.0f       // lora_alpha / r

#define BM_ 128
#define BN_ 128
#define BAUG_ 192           // BN + NR (augmented: W rows + A_flat rows)
#define BK_ 32
#define NTHREADS_ 256
#define KSTEPS_ (D_IN_ / BK_)

#define WAUG_ROWS_ 1088     // D_OUT + NR : bf16 [1088][1024] in workspace

// ---- shared memory carve (TDM kernel): 50 KB static ----
#define XS_OFF_    0                      // 8 KB   X tile bf16 (single buffer)
#define BS_OFF_    8192                   // 2x12KB W|A tile bf16 (double buffer)
#define BS_BYTES_  12288                  //        (128*64B W rows + 64*64B A rows)
#define SM_OFF_    0                      // 16 KB  masked s (reuses Xs+Bs0, epilogue only)
#define BF_OFF_    32768                  // 16 KB  B_flat tile bf16
#define SCALE_OFF_ 49152                  // 2 KB   route scales
#define SMEM_BYTES_ 51200

union FragBF {
    v16bf v;
    uint4 u[2];
};

__device__ __forceinline__ unsigned short f2bf(float f) {
    __bf16 h = (__bf16)f;
    return __builtin_bit_cast(unsigned short, h);
}
__device__ __forceinline__ unsigned int pack2(float a, float b) {
    return (unsigned int)f2bf(a) | ((unsigned int)f2bf(b) << 16);
}

__device__ __forceinline__ void async_load_b128(unsigned lds_off, const void* gptr) {
    asm volatile("global_load_async_to_lds_b128 %0, %1, off"
                 :: "v"(lds_off), "v"((unsigned long long)(size_t)gptr)
                 : "memory");
}
__device__ __forceinline__ void wait_async0() {
    asm volatile("s_wait_asynccnt 0" ::: "memory");
}

// ---- Tensor Data Mover: D# group1 for a 2D bf16 tile ----
__device__ __forceinline__ v8i tdm_g1(int tensor_d0, int tensor_d1,
                                      int tile_d0, int tile_d1, int stride0) {
    v8i g;
    g[0] = 1 << 16;                                   // data_size=1 (2 bytes), mask=0
    g[1] = (tensor_d0 & 0xffff) << 16;                // tensor_dim0[15:0] @ bits 63:48
    g[2] = ((unsigned)tensor_d0 >> 16) |              // tensor_dim0[31:16]
           ((tensor_d1 & 0xffff) << 16);              // tensor_dim1[15:0]
    g[3] = ((unsigned)tensor_d1 >> 16) |              // tensor_dim1[31:16]
           (tile_d0 << 16);                           // tile_dim0 @ bits 127:112
    g[4] = tile_d1 & 0xffff;                          // tile_dim1 (tile_dim2 = 0)
    g[5] = stride0;                                   // tensor_dim0_stride[31:0]
    g[6] = 0;
    g[7] = 0;
    return g;
}
__device__ __forceinline__ void tdm_load_2d(unsigned lds_off, unsigned long long ga, v8i g1) {
    v4u g0;
    g0.x = 1u;                                        // count=1
    g0.y = lds_off;                                   // lds_addr
    g0.z = (unsigned)ga;                              // global_addr[31:0]
    g0.w = (unsigned)(ga >> 32) | 0x80000000u;        // global_addr[56:32] | type=2
    v4i z4 = {0, 0, 0, 0};
    v8i z8 = {0, 0, 0, 0, 0, 0, 0, 0};
    __builtin_amdgcn_tensor_load_to_lds(g0, g1, z4, z4, z8, 0);
}

// ---------------- prep: fp32 -> bf16 weight copies in workspace ----------------
// Waug bf16 [1088][1024]: rows 0..1023 = W, rows 1024..1087 = A_flat
// Bfw  bf16 [1024][64]  : Bfw[n][a*16+r] = loraB[a][n][r]
#define WAUG_QUADS_ 278528   // 1088*1024/4
#define BFW_QUADS_  16384    // 1024*64/4
#define PREP_BLOCKS_ 1152    // (WAUG_QUADS_+BFW_QUADS_)/256

__global__ __launch_bounds__(NTHREADS_)
void prep_bf16(const float* __restrict__ W,
               const float* __restrict__ loraA,
               const float* __restrict__ loraB,
               uint2* __restrict__ waug,
               uint2* __restrict__ bfw) {
    int q = blockIdx.x * NTHREADS_ + threadIdx.x;
    if (q < WAUG_QUADS_) {
        int r = q >> 8, c4 = q & 255;
        const float* src = (r < D_OUT_) ? (W + (size_t)r * D_IN_)
                                        : (loraA + (size_t)(r - D_OUT_) * D_IN_);
        float4 v = *(const float4*)(src + c4 * 4);
        waug[q] = make_uint2(pack2(v.x, v.y), pack2(v.z, v.w));
    } else {
        int q2 = q - WAUG_QUADS_;
        int n = q2 >> 4, c4 = q2 & 15;
        int a = c4 >> 2, r0 = (c4 & 3) * 4;
        float4 v = *(const float4*)(loraB + (size_t)a * D_OUT_ * R_LORA_ +
                                    (size_t)n * R_LORA_ + r0);
        bfw[q2] = make_uint2(pack2(v.x, v.y), pack2(v.z, v.w));
    }
}

// ---------------- main fused kernel: TDM weights + async B_flat + reg-prefetched X ----------------
__global__ __launch_bounds__(NTHREADS_)
void lora_fused_gemm_tdm(const float* __restrict__ x,
                         const float* __restrict__ bias,
                         const float* __restrict__ masks,
                         const unsigned short* __restrict__ Waug,
                         const unsigned short* __restrict__ Bfw,
                         float* __restrict__ out) {
    __shared__ __align__(16) unsigned char smem[SMEM_BYTES_];
    unsigned short* Xs = (unsigned short*)(smem + XS_OFF_);
    float* ScaleL = (float*)(smem + SCALE_OFF_);

    const int tid  = threadIdx.x;
    const int lane = tid & 31;
    const int warp = tid >> 5;
    const int wm   = warp >> 2;     // 0..1
    const int wn   = warp & 3;      // 0..3
    const int lr   = lane & 15;
    const int hi   = lane >> 4;

    const int m0 = blockIdx.x * BM_;
    const int n0 = blockIdx.y * BN_;

    // B_flat tile via async-LDS copy (in flight across the whole main loop)
#pragma unroll
    for (int i = 0; i < 4; ++i) {
        int q = tid + NTHREADS_ * i;             // 0..1023
        int n = q >> 3, c = q & 7;
        async_load_b128(BF_OFF_ + n * 128 + c * 16,
                        (const char*)Bfw + (size_t)(n0 + n) * 128 + c * 16);
    }

    // route scales
#pragma unroll
    for (int i = 0; i < 2; ++i) {
        int idx = tid + NTHREADS_ * i;
        int row = idx >> 2, a = idx & 3;
        ScaleL[idx] = SCALING_ * masks[(size_t)a * M_TOT_ + m0 + row];
    }

    v8f acc[12];
    {
        v8f z = {0.f, 0.f, 0.f, 0.f, 0.f, 0.f, 0.f, 0.f};
#pragma unroll
        for (int i = 0; i < 12; ++i) acc[i] = z;
    }

    // X tile: per-thread strength-reduced addressing
    //   quad i covers row (tid>>3)+32*i, float4 column tid&7
    const float* xb = x + (size_t)(m0 + (tid >> 3)) * D_IN_ + (tid & 7) * 4;
    const int xsbase = (tid >> 3) * BK_ + (tid & 7) * 4;    // ushort index in Xs
    float4 xv[4];
#pragma unroll
    for (int i = 0; i < 4; ++i)
        xv[i] = *(const float4*)(xb + (size_t)i * 32 * D_IN_);

    // TDM descriptors for W|A tiles (uniform)
    const v8i g1W = tdm_g1(D_IN_, D_OUT_, BK_, BN_, D_IN_);   // 32x128 tile of W rows
    const v8i g1A = tdm_g1(D_IN_, NR_,    BK_, NR_, D_IN_);   // 32x64  tile of A rows
    const unsigned long long gaW0 = (unsigned long long)(size_t)Waug + (size_t)n0 * (D_IN_ * 2);
    const unsigned long long gaA0 = (unsigned long long)(size_t)Waug + (size_t)D_OUT_ * (D_IN_ * 2);

    if (warp == 0) {
        tdm_load_2d(BS_OFF_,        gaW0, g1W);
        tdm_load_2d(BS_OFF_ + 8192, gaA0, g1A);
    }

    for (int kidx = 0; kidx < KSTEPS_; ++kidx) {
        const int cur = kidx & 1;
        // commit X regs -> LDS (bf16)
#pragma unroll
        for (int i = 0; i < 4; ++i) {
            uint2 p = make_uint2(pack2(xv[i].x, xv[i].y), pack2(xv[i].z, xv[i].w));
            *(uint2*)&Xs[xsbase + i * (32 * BK_)] = p;
        }
        if (warp == 0) __builtin_amdgcn_s_wait_tensorcnt(0);
        __syncthreads();

        if (kidx + 1 < KSTEPS_) {
            const unsigned long long koff = (unsigned long long)(kidx + 1) * (BK_ * 2);
            if (warp == 0) {
                const unsigned bs = BS_OFF_ + (1 - cur) * BS_BYTES_;
                tdm_load_2d(bs,        gaW0 + koff, g1W);
                tdm_load_2d(bs + 8192, gaA0 + koff, g1A);
            }
            xb += BK_;
#pragma unroll
            for (int i = 0; i < 4; ++i)
                xv[i] = *(const float4*)(xb + (size_t)i * 32 * D_IN_);
        }

        const char* XsB = (const char*)Xs;
        const char* BsB = (const char*)(smem + BS_OFF_ + cur * BS_BYTES_);
        FragBF afr[4], bfr[3];
#pragma unroll
        for (int mt = 0; mt < 4; ++mt) {
            int row = (wm * 4 + mt) * 16 + lr;
            int o = row * (BK_ * 2) + hi * 32;
            afr[mt].u[0] = *(const uint4*)(XsB + o);
            afr[mt].u[1] = *(const uint4*)(XsB + o + 16);
        }
#pragma unroll
        for (int ct = 0; ct < 3; ++ct) {
            int col = (wn * 3 + ct) * 16 + lr;
            int o = col * (BK_ * 2) + hi * 32;
            bfr[ct].u[0] = *(const uint4*)(BsB + o);
            bfr[ct].u[1] = *(const uint4*)(BsB + o + 16);
        }
#pragma unroll
        for (int mt = 0; mt < 4; ++mt)
#pragma unroll
            for (int ct = 0; ct < 3; ++ct)
                acc[mt * 3 + ct] = __builtin_amdgcn_wmma_f32_16x16x32_bf16(
                    false, afr[mt].v, false, bfr[ct].v,
                    (short)0, acc[mt * 3 + ct], false, false);
        __syncthreads();
    }

    // mask + scale s tiles into Sm (overlays Xs/Bs: safe after final barrier)
    unsigned short* Sm = (unsigned short*)(smem + SM_OFF_);
#pragma unroll
    for (int mt = 0; mt < 4; ++mt) {
#pragma unroll
        for (int ct = 0; ct < 3; ++ct) {
            int gct = wn * 3 + ct;
            if (gct >= 8) {                      // wave-uniform
                int a = gct - 8;
#pragma unroll
                for (int g = 0; g < 8; ++g) {
                    int ml = (wm * 4 + mt) * 16 + g + 8 * hi;
                    float v = acc[mt * 3 + ct][g] * ScaleL[ml * 4 + a];
                    Sm[ml * NR_ + a * 16 + lr] = f2bf(v);
                }
            }
        }
    }
    wait_async0();          // B_flat async copy complete (per wave) before use
    __syncthreads();

    const char* SmB = (const char*)Sm;
    const char* BfB = (const char*)(smem + BF_OFF_);

#pragma unroll
    for (int mt = 0; mt < 4; ++mt) {
#pragma unroll
        for (int ct = 0; ct < 3; ++ct) {
            int gct = wn * 3 + ct;
            if (gct < 8) {                       // wave-uniform
                v8f c = acc[mt * 3 + ct];
#pragma unroll
                for (int ks = 0; ks < 2; ++ks) {
                    FragBF af, bg;
                    int row = (wm * 4 + mt) * 16 + lr;
                    int oa = row * (NR_ * 2) + ks * 64 + hi * 32;
                    af.u[0] = *(const uint4*)(SmB + oa);
                    af.u[1] = *(const uint4*)(SmB + oa + 16);
                    int col = gct * 16 + lr;
                    int ob = col * (NR_ * 2) + ks * 64 + hi * 32;
                    bg.u[0] = *(const uint4*)(BfB + ob);
                    bg.u[1] = *(const uint4*)(BfB + ob + 16);
                    c = __builtin_amdgcn_wmma_f32_16x16x32_bf16(
                        false, af.v, false, bg.v, (short)0, c, false, false);
                }
                int ng = n0 + gct * 16 + lr;
                float bv = bias[ng];
#pragma unroll
                for (int g = 0; g < 8; ++g) {
                    int mg = m0 + (wm * 4 + mt) * 16 + g + 8 * hi;
                    out[(size_t)mg * D_OUT_ + ng] = c[g] + bv;
                }
            }
        }
    }
}

// ---------------- fallback: inline conversion, no workspace ----------------
__global__ __launch_bounds__(NTHREADS_)
void lora_fused_gemm(const float* __restrict__ x,
                     const float* __restrict__ W,
                     const float* __restrict__ bias,
                     const float* __restrict__ loraA,
                     const float* __restrict__ loraB,
                     const float* __restrict__ masks,
                     float* __restrict__ out) {
    __shared__ unsigned short Xs[BM_ * BK_];
    __shared__ unsigned short Bs[BAUG_ * BK_];
    __shared__ unsigned short Bf[BN_ * NR_];
    __shared__ unsigned short Sm[BM_ * NR_];
    __shared__ float ScaleL[BM_ * N_ADAPT_];

    const int tid  = threadIdx.x;
    const int lane = tid & 31;
    const int warp = tid >> 5;
    const int wm   = warp >> 2;
    const int wn   = warp & 3;
    const int lr   = lane & 15;
    const int hi   = lane >> 4;

    const int m0 = blockIdx.x * BM_;
    const int n0 = blockIdx.y * BN_;

#pragma unroll
    for (int i = 0; i < 2; ++i) {
        int idx = tid + NTHREADS_ * i;
        int row = idx >> 2, a = idx & 3;
        ScaleL[idx] = SCALING_ * masks[(size_t)a * M_TOT_ + m0 + row];
    }

    v8f acc[12];
    {
        v8f z = {0.f, 0.f, 0.f, 0.f, 0.f, 0.f, 0.f, 0.f};
#pragma unroll
        for (int i = 0; i < 12; ++i) acc[i] = z;
    }

    const char* XsB = (const char*)Xs;
    const char* BsB = (const char*)Bs;

    for (int k0 = 0; k0 < D_IN_; k0 += BK_) {
        __syncthreads();
#pragma unroll
        for (int i = 0; i < 4; ++i) {
            int q = tid + NTHREADS_ * i;
            int row = q >> 3, c4 = q & 7;
            float4 v = *(const float4*)(x + (size_t)(m0 + row) * D_IN_ + k0 + c4 * 4);
            uint2 p = make_uint2(pack2(v.x, v.y), pack2(v.z, v.w));
            *(uint2*)&Xs[row * BK_ + c4 * 4] = p;
        }
#pragma unroll
        for (int i = 0; i < 6; ++i) {
            int q = tid + NTHREADS_ * i;
            int r = q >> 3, c4 = q & 7;
            const float* src = (r < BN_) ? (W + (size_t)(n0 + r) * D_IN_)
                                         : (loraA + (size_t)(r - BN_) * D_IN_);
            float4 v = *(const float4*)(src + k0 + c4 * 4);
            uint2 p = make_uint2(pack2(v.x, v.y), pack2(v.z, v.w));
            *(uint2*)&Bs[r * BK_ + c4 * 4] = p;
        }
        __syncthreads();

        FragBF afr[4], bfr[3];
#pragma unroll
        for (int mt = 0; mt < 4; ++mt) {
            int row = (wm * 4 + mt) * 16 + lr;
            int o = row * (BK_ * 2) + hi * 32;
            afr[mt].u[0] = *(const uint4*)(XsB + o);
            afr[mt].u[1] = *(const uint4*)(XsB + o + 16);
        }
#pragma unroll
        for (int ct = 0; ct < 3; ++ct) {
            int col = (wn * 3 + ct) * 16 + lr;
            int o = col * (BK_ * 2) + hi * 32;
            bfr[ct].u[0] = *(const uint4*)(BsB + o);
            bfr[ct].u[1] = *(const uint4*)(BsB + o + 16);
        }
#pragma unroll
        for (int mt = 0; mt < 4; ++mt)
#pragma unroll
            for (int ct = 0; ct < 3; ++ct)
                acc[mt * 3 + ct] = __builtin_amdgcn_wmma_f32_16x16x32_bf16(
                    false, afr[mt].v, false, bfr[ct].v,
                    (short)0, acc[mt * 3 + ct], false, false);
    }

#pragma unroll
    for (int i = 0; i < 8; ++i) {
        int q = tid + NTHREADS_ * i;
        int n = q >> 4, c4 = q & 15;
        int a = c4 >> 2, r0 = (c4 & 3) * 4;
        float4 v = *(const float4*)(loraB + (size_t)a * D_OUT_ * R_LORA_ +
                                    (size_t)(n0 + n) * R_LORA_ + r0);
        uint2 p = make_uint2(pack2(v.x, v.y), pack2(v.z, v.w));
        *(uint2*)&Bf[n * NR_ + c4 * 4] = p;
    }

#pragma unroll
    for (int mt = 0; mt < 4; ++mt) {
#pragma unroll
        for (int ct = 0; ct < 3; ++ct) {
            int gct = wn * 3 + ct;
            if (gct >= 8) {
                int a = gct - 8;
#pragma unroll
                for (int g = 0; g < 8; ++g) {
                    int ml = (wm * 4 + mt) * 16 + g + 8 * hi;
                    float v = acc[mt * 3 + ct][g] * ScaleL[ml * 4 + a];
                    Sm[ml * NR_ + a * 16 + lr] = f2bf(v);
                }
            }
        }
    }
    __syncthreads();

    const char* SmB = (const char*)Sm;
    const char* BfB = (const char*)Bf;

#pragma unroll
    for (int mt = 0; mt < 4; ++mt) {
#pragma unroll
        for (int ct = 0; ct < 3; ++ct) {
            int gct = wn * 3 + ct;
            if (gct < 8) {
                v8f c = acc[mt * 3 + ct];
#pragma unroll
                for (int ks = 0; ks < 2; ++ks) {
                    FragBF af, bg;
                    int row = (wm * 4 + mt) * 16 + lr;
                    int oa = row * (NR_ * 2) + ks * 64 + hi * 32;
                    af.u[0] = *(const uint4*)(SmB + oa);
                    af.u[1] = *(const uint4*)(SmB + oa + 16);
                    int col = gct * 16 + lr;
                    int ob = col * (NR_ * 2) + ks * 64 + hi * 32;
                    bg.u[0] = *(const uint4*)(BfB + ob);
                    bg.u[1] = *(const uint4*)(BfB + ob + 16);
                    c = __builtin_amdgcn_wmma_f32_16x16x32_bf16(
                        false, af.v, false, bg.v, (short)0, c, false, false);
                }
                int ng = n0 + gct * 16 + lr;
                float bv = bias[ng];
#pragma unroll
                for (int g = 0; g < 8; ++g) {
                    int mg = m0 + (wm * 4 + mt) * 16 + g + 8 * hi;
                    out[(size_t)mg * D_OUT_ + ng] = c[g] + bv;
                }
            }
        }
    }
}

extern "C" void kernel_launch(void* const* d_in, const int* in_sizes, int n_in,
                              void* d_out, int out_size, void* d_ws, size_t ws_size,
                              hipStream_t stream) {
    const float* x     = (const float*)d_in[0];
    const float* W     = (const float*)d_in[1];
    const float* b     = (const float*)d_in[2];
    const float* loraA = (const float*)d_in[3];
    const float* loraB = (const float*)d_in[4];
    const float* masks = (const float*)d_in[5];
    float* out = (float*)d_out;

    dim3 grid(M_TOT_ / BM_, D_OUT_ / BN_);   // 512 x 8

    const size_t needWS = (size_t)(WAUG_ROWS_ * D_IN_ + D_OUT_ * NR_) * 2;  // ~2.3 MB
    if (ws_size >= needWS) {
        unsigned short* Waug = (unsigned short*)d_ws;
        unsigned short* Bfw  = Waug + (size_t)WAUG_ROWS_ * D_IN_;
        prep_bf16<<<PREP_BLOCKS_, NTHREADS_, 0, stream>>>(W, loraA, loraB,
                                                          (uint2*)Waug, (uint2*)Bfw);
        lora_fused_gemm_tdm<<<grid, NTHREADS_, 0, stream>>>(x, b, masks, Waug, Bfw, out);
    } else {
        lora_fused_gemm<<<grid, NTHREADS_, 0, stream>>>(x, W, b, loraA, loraB, masks, out);
    }
}
